// CRNN_26525718020754
// MI455X (gfx1250) — compile-verified, matrix operand order checked by reading, not run on hardware
//
#include <hip/hip_runtime.h>

// ---------------------------------------------------------------------------
// Types & WMMA helpers (gfx1250: wave32, v_wmma_f32_16x16x32_bf16)
// ---------------------------------------------------------------------------
typedef __bf16 bf16;
typedef __attribute__((ext_vector_type(16))) __bf16 bf16x16;
typedef __attribute__((ext_vector_type(8)))  __bf16 bf16x8;
typedef __attribute__((ext_vector_type(8)))  float  f32x8;

__device__ __forceinline__ bf16 f2bf(float f) {
    union { float f; unsigned u; } v; v.f = f;
    unsigned r = v.u + 0x7FFFu + ((v.u >> 16) & 1u);
    union { unsigned short s; bf16 b; } o; o.s = (unsigned short)(r >> 16);
    return o.b;
}
__device__ __forceinline__ float bf2f(bf16 b) {
    union { unsigned short s; bf16 b; } i; i.b = b;
    union { unsigned u; float f; } o; o.u = ((unsigned)i.s) << 16;
    return o.f;
}
__device__ __forceinline__ f32x8 wmma_bf16(bf16x16 a, bf16x16 b, f32x8 c) {
    return __builtin_amdgcn_wmma_f32_16x16x32_bf16(false, a, false, b, (short)0, c,
                                                   false, false);
}
// A/B fragment K index for element e (0..15) of the v16bf (ISA 7.12.2 layout)
__device__ __forceinline__ int frag_k(int lane, int e) {
    int kh = ((lane >> 4) & 1) * 8;
    return (e < 8) ? (kh + e) : (16 + kh + (e - 8));
}
// B fragment (row n of [N,K] row-major bf16, contiguous K) via two 16B loads
__device__ __forceinline__ bf16x16 glb_frag(const bf16* __restrict__ B, int n, int ldb,
                                            int k0, int lane) {
    int khalf = (lane >> 4) & 1;
    const bf16* p = B + (size_t)n * ldb + k0 + khalf * 8;
    bf16x8 lo = *(const bf16x8*)p;
    bf16x8 hi = *(const bf16x8*)(p + 16);
    bf16x16 r;
#pragma unroll
    for (int i = 0; i < 8; ++i) { r[i] = lo[i]; r[8 + i] = hi[i]; }
    return r;
}
// A/B fragment from LDS (row-major bf16 with leading dim ld)
__device__ __forceinline__ bf16x16 lds_frag(const bf16* base, int row, int ld, int k0,
                                            int lane) {
    int khalf = (lane >> 4) & 1;
    const bf16* p = base + row * ld + k0 + khalf * 8;
    bf16x16 a;
#pragma unroll
    for (int i = 0; i < 8; ++i) { a[i] = p[i]; a[8 + i] = p[16 + i]; }
    return a;
}
__device__ __forceinline__ float sigm(float x) { return 1.f / (1.f + __expf(-x)); }

// ---- CDNA5 async global->LDS (ASYNCcnt path, ISA ch10 / 15.18.3) ----------
__device__ __forceinline__ void async_load_b128(unsigned lds_off, const void* gaddr) {
    asm volatile("global_load_async_to_lds_b128 %0, %1, off"
                 :: "v"(lds_off), "v"(gaddr) : "memory");
}
__device__ __forceinline__ void wait_asynccnt0() {
    asm volatile("s_wait_asynccnt 0x0" ::: "memory");
}

// ---------------------------------------------------------------------------
// Elementwise prep kernels
// ---------------------------------------------------------------------------
__global__ void k_f32_to_bf16(const float* __restrict__ s, bf16* __restrict__ d, int n) {
    int i = blockIdx.x * blockDim.x + threadIdx.x;
    if (i < n) d[i] = f2bf(s[i]);
}
__global__ void k_vec_add(const float* __restrict__ a, const float* __restrict__ b,
                          float* __restrict__ o, int n) {
    int i = blockIdx.x * blockDim.x + threadIdx.x;
    if (i < n) o[i] = a[i] + b[i];
}
// conv weight (O,I,KH,KW) f32 -> bf16 [O,Kpad], k = (kh*KW+kw)*I + ci, zero padded
__global__ void k_wprep(const float* __restrict__ w, bf16* __restrict__ wk, int O, int I,
                        int KH, int KW, int Kpad) {
    int idx = blockIdx.x * blockDim.x + threadIdx.x;
    if (idx >= O * Kpad) return;
    int o = idx / Kpad, k = idx - o * Kpad;
    float v = 0.f;
    int Kreal = KH * KW * I;
    if (k < Kreal) {
        int ci = k % I, pix = k / I;
        int kw = pix % KW, kh = pix / KW;
        v = w[(((size_t)o * I + ci) * KH + kh) * KW + kw];
    }
    wk[idx] = f2bf(v);
}
// per-channel scale/shift for conv epilogue (folds bias + eval-mode BN)
__global__ void k_ssprep(const float* __restrict__ bias, const float* __restrict__ g,
                         const float* __restrict__ bnb, const float* __restrict__ m,
                         const float* __restrict__ v, float* __restrict__ sc,
                         float* __restrict__ sh, int O, int has_bn) {
    int o = blockIdx.x * blockDim.x + threadIdx.x;
    if (o >= O) return;
    if (has_bn) {
        float s = g[o] * rsqrtf(v[o] + 1e-5f);
        sc[o] = s;
        sh[o] = (bias[o] - m[o]) * s + bnb[o];
    } else {
        sc[o] = 1.f;
        sh[o] = bias[o];
    }
}

// ---------------------------------------------------------------------------
// GEMM: C[M,N](f32,ldc) = A[M,K](bf16,lda) * B[N,K](bf16,ldb)^T + bias.
// Block = 4 waves = 4 m-tiles x one 64-col group; wave computes 16x64 (4 accs,
// one A frag per k-step feeds 4 WMMAs). The shared 64x32 B panel is double-
// buffered in LDS via global_load_async_to_lds_b128 (ASYNCcnt + barrier).
// M%16==0, K%32==0 required; N arbitrary (stores masked; B over-read stays in ws).
// ---------------------------------------------------------------------------
__global__ void __launch_bounds__(128) k_gemm(const bf16* __restrict__ A, int lda,
                                              const bf16* __restrict__ Bm, int ldb,
                                              const float* __restrict__ bias,
                                              float* __restrict__ C, int ldc, int M,
                                              int N, int K, int relu) {
    __shared__ bf16 sB[2][64 * 32];  // 8 KB double buffer
    const int tid = threadIdx.x, lane = tid & 31, wave = tid >> 5;
    const int mtiles = M >> 4;
    const int ngroups = (N + 63) >> 6;
    int mb = blockIdx.x / ngroups, ng = blockIdx.x - mb * ngroups;
    int mt = mb * 4 + wave;
    const bool active = (mt < mtiles);
    if (!active) mt = mtiles - 1;  // clamp: keep barriers/EXEC uniform, mask stores
    const int n0 = ng << 6;
    const int arow = (mt << 4) + (lane & 15);
    const int khalf = (lane >> 4) & 1;

    auto stage = [&](int buf, int k0) {
#pragma unroll
        for (int it = 0; it < 2; ++it) {
            int chunk = tid + it * 128;           // 256 x 16B = 64 rows x 64B
            int row = chunk >> 2, part = chunk & 3;
            const bf16* g = Bm + (size_t)(n0 + row) * ldb + k0 + part * 8;
            unsigned l = (unsigned)(size_t)&sB[buf][row * 32 + part * 8];
            async_load_b128(l, g);
        }
    };

    f32x8 acc[4];
#pragma unroll
    for (int j = 0; j < 4; ++j)
#pragma unroll
        for (int v = 0; v < 8; ++v) acc[j][v] = 0.f;

    stage(0, 0);
    int cur = 0;
    for (int k0 = 0; k0 < K; k0 += 32, cur ^= 1) {
        wait_asynccnt0();    // this wave's async stores to LDS landed
        __syncthreads();     // everyone's landed; prev-iter readers all done
        if (k0 + 32 < K) stage(cur ^ 1, k0 + 32);
        const bf16* p = A + (size_t)arow * lda + k0 + khalf * 8;
        if (k0 + 64 < K) __builtin_prefetch(p + 64, 0, 1);
        bf16x16 a;
        {
            bf16x8 lo = *(const bf16x8*)p;
            bf16x8 hi = *(const bf16x8*)(p + 16);
#pragma unroll
            for (int i = 0; i < 8; ++i) { a[i] = lo[i]; a[8 + i] = hi[i]; }
        }
#pragma unroll
        for (int j = 0; j < 4; ++j) {
            bf16x16 b = lds_frag(sB[cur], (j << 4) + (lane & 15), 32, 0, lane);
            acc[j] = wmma_bf16(a, b, acc[j]);
        }
    }
    if (active) {
#pragma unroll
        for (int j = 0; j < 4; ++j)
#pragma unroll
            for (int v = 0; v < 8; ++v) {
                int row = (mt << 4) + ((lane >> 4) << 3) + v;
                int col = n0 + (j << 4) + (lane & 15);
                if (col < N) {
                    float val = acc[j][v] + (bias ? bias[col] : 0.f);
                    if (relu) val = fmaxf(val, 0.f);
                    C[(size_t)row * ldc + col] = val;
                }
            }
    }
}

// ---------------------------------------------------------------------------
// Implicit-GEMM conv (NHWC bf16 in -> NHWC bf16 out, fused scale/shift + ReLU)
// M = 32*OH*OW positions, N = O channels, K = KH*KW*C (padded to Kpad%32==0).
// For C%8==0 each 8-element fragment half is one pixel-contiguous 16B load.
// ---------------------------------------------------------------------------
__global__ void __launch_bounds__(128) k_conv(const bf16* __restrict__ in,
                                              const bf16* __restrict__ wk,
                                              const float* __restrict__ sc,
                                              const float* __restrict__ sh,
                                              bf16* __restrict__ out, int H, int W,
                                              int C, int OH, int OW, int O, int KH,
                                              int KW, int pad, int Kpad, int Kreal) {
    const int lane = threadIdx.x & 31, wave = threadIdx.x >> 5;
    const int ntiles = O >> 4;
    const int M = 32 * OH * OW;
    const int mtiles = M >> 4;
    int tile = blockIdx.x * 4 + wave;
    if (tile >= mtiles * ntiles) return;
    int mt = tile / ntiles, nt = tile - mt * ntiles;
    int m = (mt << 4) + (lane & 15);
    int bb = m / (OH * OW);
    int rem = m - bb * (OH * OW);
    int oh = rem / OW, ow = rem - (rem / OW) * OW;
    int ncol = (nt << 4) + (lane & 15);
    const int khalf = (lane >> 4) & 1;
    bf16 zb = f2bf(0.f);
    f32x8 acc;
#pragma unroll
    for (int v = 0; v < 8; ++v) acc[v] = 0.f;
    const bool vec_ok = ((C & 7) == 0);
    for (int k0 = 0; k0 < Kpad; k0 += 32) {
        bf16x16 a;
        if (vec_ok) {
            // two pixel-contiguous 8-wide runs per lane
            bf16 tmp[16];
#pragma unroll
            for (int half = 0; half < 2; ++half) {
                int kb = k0 + half * 16 + khalf * 8;
                bool ld = false;
                size_t base = 0;
                if (kb < Kreal) {
                    int ci = kb % C, pix = kb / C;
                    int kw = pix % KW, kh = pix / KW;
                    int ih = oh + kh - pad, iw = ow + kw - pad;
                    if (ih >= 0 && ih < H && iw >= 0 && iw < W) {
                        ld = true;
                        base = (((size_t)bb * H + ih) * W + iw) * C + ci;
                    }
                }
                if (ld) {
                    bf16x8 v = *(const bf16x8*)&in[base];
#pragma unroll
                    for (int i = 0; i < 8; ++i) tmp[half * 8 + i] = v[i];
                } else {
#pragma unroll
                    for (int i = 0; i < 8; ++i) tmp[half * 8 + i] = zb;
                }
            }
#pragma unroll
            for (int i = 0; i < 16; ++i) a[i] = tmp[i];
        } else {
#pragma unroll
            for (int e = 0; e < 16; ++e) {
                int k = k0 + frag_k(lane, e);
                bf16 val = zb;
                if (k < Kreal) {
                    int ci = k % C, pix = k / C;
                    int kw = pix % KW, kh = pix / KW;
                    int ih = oh + kh - pad, iw = ow + kw - pad;
                    if (ih >= 0 && ih < H && iw >= 0 && iw < W)
                        val = in[(((size_t)bb * H + ih) * W + iw) * C + ci];
                }
                a[e] = val;
            }
        }
        bf16x16 b = glb_frag(wk, ncol, Kpad, k0, lane);
        acc = wmma_bf16(a, b, acc);
    }
#pragma unroll
    for (int v = 0; v < 8; ++v) {
        int row = (mt << 4) + ((lane >> 4) << 3) + v;
        int col = (nt << 4) + (lane & 15);
        float val = fmaxf(acc[v] * sc[col] + sh[col], 0.f);
        out[(size_t)row * O + col] = f2bf(val);
    }
}

// 2x2 max pool, NHWC bf16, -inf padding semantics
__global__ void k_pool(const bf16* __restrict__ in, bf16* __restrict__ out, int H, int W,
                       int C, int OH, int OW, int sh, int sw, int ph, int pw) {
    int idx = blockIdx.x * blockDim.x + threadIdx.x;
    int n = 32 * OH * OW * C;
    if (idx >= n) return;
    int c = idx % C, r = idx / C;
    int ow = r % OW; r /= OW;
    int oh = r % OH; int b = r / OH;
    float best = -3.4e38f;
    for (int kh = 0; kh < 2; ++kh)
        for (int kw = 0; kw < 2; ++kw) {
            int ih = oh * sh - ph + kh, iw = ow * sw - pw + kw;
            if (ih >= 0 && ih < H && iw >= 0 && iw < W)
                best = fmaxf(best, bf2f(in[(((size_t)b * H + ih) * W + iw) * C + c]));
        }
    out[idx] = f2bf(best);
}

// conv feature map [B,1,T,512] (NHWC) -> sequence x1 [T*B, 512] bf16
__global__ void k_to_seq(const bf16* __restrict__ a6, bf16* __restrict__ x1, int T) {
    int idx = blockIdx.x * blockDim.x + threadIdx.x;
    int n = T * 32 * 512;
    if (idx >= n) return;
    int c = idx & 511;
    int r = idx >> 9;
    int b = r & 31;
    int t = r >> 5;
    x1[idx] = a6[((size_t)b * T + t) * 512 + c];
}

// ---------------------------------------------------------------------------
// Persistent BiLSTM recurrence. grid=2 (dir), block=1024 (32 waves).
// Wave w owns (mtile=w&1, hchunk=w>>1): computes i/f/g/o gate tiles with WMMA,
// keeps cell state c in registers (f32x8), h lives in LDS as bf16 (next A frag).
// xproj already holds x@Wih^T + bih + bhh. Writes h into hs_cat at col offset.
// ---------------------------------------------------------------------------
__global__ void __launch_bounds__(1024) k_lstm(const float* __restrict__ xp_f,
                                               const float* __restrict__ xp_r,
                                               const bf16* __restrict__ whh_f,
                                               const bf16* __restrict__ whh_r,
                                               bf16* __restrict__ hs_cat, int T) {
    __shared__ bf16 sh_h[32 * 256];
    const int dir = blockIdx.x;
    const float* xp = dir ? xp_r : xp_f;
    const bf16* whh = dir ? whh_r : whh_f;
    const int coff = dir * 256;
    const int tid = threadIdx.x, lane = tid & 31, wave = tid >> 5;
    const int mt = wave & 1, h0 = (wave >> 1) << 4;
    for (int i = tid; i < 32 * 256; i += 1024) sh_h[i] = f2bf(0.f);
    f32x8 cst;
#pragma unroll
    for (int v = 0; v < 8; ++v) cst[v] = 0.f;
    __syncthreads();
    const int arow = (mt << 4) + (lane & 15);
    const int ncol = h0 + (lane & 15);
    for (int it = 0; it < T; ++it) {
        int t = dir ? (T - 1 - it) : it;
        f32x8 gi, gf, gg, go;
#pragma unroll
        for (int v = 0; v < 8; ++v) {
            int b = (mt << 4) + ((lane >> 4) << 3) + v;
            size_t ro = ((size_t)t * 32 + b) * 1024;
            gi[v] = xp[ro + 0 + ncol];
            gf[v] = xp[ro + 256 + ncol];
            gg[v] = xp[ro + 512 + ncol];
            go[v] = xp[ro + 768 + ncol];
        }
        for (int k0 = 0; k0 < 256; k0 += 32) {
            bf16x16 a = lds_frag(sh_h, arow, 256, k0, lane);
            gi = wmma_bf16(a, glb_frag(whh, ncol, 256, k0, lane), gi);
            gf = wmma_bf16(a, glb_frag(whh, 256 + ncol, 256, k0, lane), gf);
            gg = wmma_bf16(a, glb_frag(whh, 512 + ncol, 256, k0, lane), gg);
            go = wmma_bf16(a, glb_frag(whh, 768 + ncol, 256, k0, lane), go);
        }
        float hnew[8];
#pragma unroll
        for (int v = 0; v < 8; ++v) {
            float c = sigm(gf[v]) * cst[v] + sigm(gi[v]) * tanhf(gg[v]);
            cst[v] = c;
            hnew[v] = sigm(go[v]) * tanhf(c);
        }
        __syncthreads();  // all waves done reading sh_h
#pragma unroll
        for (int v = 0; v < 8; ++v) {
            int b = (mt << 4) + ((lane >> 4) << 3) + v;
            bf16 hb = f2bf(hnew[v]);
            sh_h[b * 256 + ncol] = hb;
            hs_cat[((size_t)t * 32 + b) * 512 + coff + ncol] = hb;
        }
        __syncthreads();  // h ready for next step
    }
}

// ---------------------------------------------------------------------------
// Persistent attention-GRU decoder. grid=1, block=1024. T=129, STEPS=24.
// ---------------------------------------------------------------------------
#define DT 129
__global__ void __launch_bounds__(1024) k_decoder(
    const float* __restrict__ fproj, const float* __restrict__ feats,
    const float* __restrict__ score, const bf16* __restrict__ h2h,
    const float* __restrict__ h2hb, const bf16* __restrict__ gwih,
    const bf16* __restrict__ gwhh, const float* __restrict__ gbih,
    const float* __restrict__ gbhh, float* __restrict__ hs_dec, int STEPS) {
    __shared__ bf16 sh_h[32 * 128];
    __shared__ bf16 sh_ctx[32 * 256];
    __shared__ float sh_hp[32 * 128];
    __shared__ float sh_e[DT * 32];
    const int tid = threadIdx.x, lane = tid & 31, wave = tid >> 5;
    const int mt = wave & 1, j0 = (wave >> 1) << 4;
    const int arow = (mt << 4) + (lane & 15);
    const int jcol = j0 + (lane & 15);
    for (int i = tid; i < 32 * 128; i += 1024) sh_h[i] = f2bf(0.f);
    __syncthreads();
    for (int s = 0; s < STEPS; ++s) {
        // hp = h @ h2h^T + b  (16 waves, one tile each)
        if (wave < 16) {
            f32x8 acc;
#pragma unroll
            for (int v = 0; v < 8; ++v) acc[v] = 0.f;
            for (int k0 = 0; k0 < 128; k0 += 32) {
                bf16x16 a = lds_frag(sh_h, arow, 128, k0, lane);
                acc = wmma_bf16(a, glb_frag(h2h, jcol, 128, k0, lane), acc);
            }
#pragma unroll
            for (int v = 0; v < 8; ++v) {
                int b = (mt << 4) + ((lane >> 4) << 3) + v;
                sh_hp[b * 128 + jcol] = acc[v] + h2hb[jcol];
            }
        }
        __syncthreads();
        // e[t,b] = sum_j tanh(fproj + hp) * score
        for (int r = tid; r < DT * 32; r += 1024) {
            int b = r & 31;
            float acc = 0.f;
            const float* fp = fproj + (size_t)r * 128;
            const float* hp = sh_hp + b * 128;
            for (int j = 0; j < 128; ++j) acc += tanhf(fp[j] + hp[j]) * score[j];
            sh_e[r] = acc;
        }
        __syncthreads();
        // softmax over t for each b
        if (tid < 32) {
            int b = tid;
            float mx = -3.4e38f;
            for (int t = 0; t < DT; ++t) mx = fmaxf(mx, sh_e[t * 32 + b]);
            float sum = 0.f;
            for (int t = 0; t < DT; ++t) {
                float e = __expf(sh_e[t * 32 + b] - mx);
                sh_e[t * 32 + b] = e;
                sum += e;
            }
            float inv = 1.f / sum;
            for (int t = 0; t < DT; ++t) sh_e[t * 32 + b] *= inv;
        }
        __syncthreads();
        // ctx[b,c] = sum_t alpha * feats
        for (int o = tid; o < 32 * 256; o += 1024) {
            int b = o >> 8, c = o & 255;
            float acc = 0.f;
            for (int t = 0; t < DT; ++t)
                acc += sh_e[t * 32 + b] * feats[((size_t)t * 32 + b) * 256 + c];
            sh_ctx[b * 256 + c] = f2bf(acc);
        }
        __syncthreads();
        // GRU: gi = ctx@Wih^T, gh = h@Whh^T, register-aligned r/z/n fragments
        f32x8 gir, giz, gin, ghr, ghz, ghn;
        if (wave < 16) {
#pragma unroll
            for (int v = 0; v < 8; ++v) {
                gir[v] = giz[v] = gin[v] = ghr[v] = ghz[v] = ghn[v] = 0.f;
            }
            for (int k0 = 0; k0 < 256; k0 += 32) {
                bf16x16 a = lds_frag(sh_ctx, arow, 256, k0, lane);
                gir = wmma_bf16(a, glb_frag(gwih, jcol, 256, k0, lane), gir);
                giz = wmma_bf16(a, glb_frag(gwih, 128 + jcol, 256, k0, lane), giz);
                gin = wmma_bf16(a, glb_frag(gwih, 256 + jcol, 256, k0, lane), gin);
            }
            for (int k0 = 0; k0 < 128; k0 += 32) {
                bf16x16 a = lds_frag(sh_h, arow, 128, k0, lane);
                ghr = wmma_bf16(a, glb_frag(gwhh, jcol, 128, k0, lane), ghr);
                ghz = wmma_bf16(a, glb_frag(gwhh, 128 + jcol, 128, k0, lane), ghz);
                ghn = wmma_bf16(a, glb_frag(gwhh, 256 + jcol, 128, k0, lane), ghn);
            }
        }
        float hnew[8];
        if (wave < 16) {
#pragma unroll
            for (int v = 0; v < 8; ++v) {
                int b = (mt << 4) + ((lane >> 4) << 3) + v;
                int j = jcol;
                float r = sigm(gir[v] + gbih[j] + ghr[v] + gbhh[j]);
                float z = sigm(giz[v] + gbih[128 + j] + ghz[v] + gbhh[128 + j]);
                float nn = tanhf(gin[v] + gbih[256 + j] + r * (ghn[v] + gbhh[256 + j]));
                float hold = bf2f(sh_h[b * 128 + j]);
                hnew[v] = (1.f - z) * nn + z * hold;
            }
        }
        __syncthreads();  // all reads of sh_h done
        if (wave < 16) {
#pragma unroll
            for (int v = 0; v < 8; ++v) {
                int b = (mt << 4) + ((lane >> 4) << 3) + v;
                sh_h[b * 128 + jcol] = f2bf(hnew[v]);
                hs_dec[((size_t)s * 32 + b) * 128 + jcol] = hnew[v];
            }
        }
        __syncthreads();
    }
}

// ---------------------------------------------------------------------------
// Ragged gather: labels ordered batch-major (b asc, t within)
// ---------------------------------------------------------------------------
__global__ void k_cum(const int* __restrict__ len, int* __restrict__ cum) {
    if (threadIdx.x == 0 && blockIdx.x == 0) {
        int s = 0;
        for (int b = 0; b < 32; ++b) { cum[b] = s; s += len[b]; }
        cum[32] = s;
    }
}
__global__ void k_gather(const float* __restrict__ hs_dec, const int* __restrict__ cum,
                         bf16* __restrict__ Agen, int total) {
    int idx = blockIdx.x * blockDim.x + threadIdx.x;
    if (idx >= total * 128) return;
    int l = idx >> 7, j = idx & 127;
    int b = 0;
    while (b < 31 && l >= cum[b + 1]) ++b;
    int t = l - cum[b];
    Agen[idx] = f2bf(hs_dec[((size_t)t * 32 + b) * 128 + j]);
}

// ---------------------------------------------------------------------------
// Host orchestration
// ---------------------------------------------------------------------------
static inline int ceil_div(int a, int b) { return (a + b - 1) / b; }

extern "C" void kernel_launch(void* const* d_in, const int* in_sizes, int n_in,
                              void* d_out, int out_size, void* d_ws, size_t ws_size,
                              hipStream_t stream) {
    (void)in_sizes; (void)n_in; (void)ws_size;
    const int T = 129, STEPS = 24;
    const int TOTAL = out_size / 1000;  // 416 labels

    // ---- input pointers (setup_inputs dict order) ----
    const float* IN_X = (const float*)d_in[0];
    const int* IN_LEN = (const int*)d_in[1];
    const float *CW[7], *CB[7];
    for (int i = 0; i < 7; ++i) { CW[i] = (const float*)d_in[2 + 2 * i]; CB[i] = (const float*)d_in[3 + 2 * i]; }
    const float* BN2[4]; const float* BN4[4]; const float* BN6[4];
    for (int i = 0; i < 4; ++i) { BN2[i] = (const float*)d_in[16 + i]; BN4[i] = (const float*)d_in[20 + i]; BN6[i] = (const float*)d_in[24 + i]; }
    const float* L1F[4]; const float* L1R[4]; const float* L2F[4]; const float* L2R[4];
    for (int i = 0; i < 4; ++i) { L1F[i] = (const float*)d_in[28 + i]; L1R[i] = (const float*)d_in[32 + i]; }
    const float* EMB1W = (const float*)d_in[36]; const float* EMB1B = (const float*)d_in[37];
    for (int i = 0; i < 4; ++i) { L2F[i] = (const float*)d_in[38 + i]; L2R[i] = (const float*)d_in[42 + i]; }
    const float* EMB2W = (const float*)d_in[46]; const float* EMB2B = (const float*)d_in[47];
    const float* I2HW = (const float*)d_in[48];
    const float* H2HW = (const float*)d_in[49]; const float* H2HB = (const float*)d_in[50];
    const float* SCOREW = (const float*)d_in[51];
    const float* GWIH = (const float*)d_in[52]; const float* GWHH = (const float*)d_in[53];
    const float* GBIH = (const float*)d_in[54]; const float* GBHH = (const float*)d_in[55];
    const float* GENW = (const float*)d_in[56]; const float* GENB = (const float*)d_in[57];

    // ---- workspace carve-up ----
    char* wsp = (char*)d_ws;
    size_t off = 0;
    auto alloc = [&](size_t bytes) -> char* {
        off = (off + 255) & ~(size_t)255;
        char* p = wsp + off;
        off += bytes;
        return p;
    };
    // conv descriptors: {H,W,C,OH,OW,O,KH,KW,pad}
    const int cd[7][9] = {
        {32, 512, 1, 32, 512, 64, 3, 3, 1},  {16, 256, 64, 16, 256, 128, 3, 3, 1},
        {8, 128, 128, 8, 128, 256, 3, 3, 1}, {8, 128, 256, 8, 128, 256, 3, 3, 1},
        {4, 129, 256, 4, 129, 512, 3, 3, 1}, {4, 129, 512, 4, 129, 512, 3, 3, 1},
        {2, 130, 512, 1, 129, 512, 2, 2, 0}};
    int Kreal[7], Kpad[7];
    bf16* wk[7]; float* csc[7]; float* csh[7];
    for (int i = 0; i < 7; ++i) {
        Kreal[i] = cd[i][6] * cd[i][7] * cd[i][2];
        Kpad[i] = (Kreal[i] + 31) & ~31;
        wk[i] = (bf16*)alloc((size_t)cd[i][5] * Kpad[i] * 2);
        csc[i] = (float*)alloc(cd[i][5] * 4);
        csh[i] = (float*)alloc(cd[i][5] * 4);
    }
    bf16* bufA = (bf16*)alloc(67108864);
    bf16* bufB = (bf16*)alloc(67108864);
    bf16* a0in = (bf16*)alloc((size_t)32 * 32 * 512 * 2);
    bf16* wih1f = (bf16*)alloc(1024 * 512 * 2); bf16* whh1f = (bf16*)alloc(1024 * 256 * 2);
    bf16* wih1r = (bf16*)alloc(1024 * 512 * 2); bf16* whh1r = (bf16*)alloc(1024 * 256 * 2);
    bf16* emb1w = (bf16*)alloc(256 * 512 * 2);
    bf16* wih2f = (bf16*)alloc(1024 * 256 * 2); bf16* whh2f = (bf16*)alloc(1024 * 256 * 2);
    bf16* wih2r = (bf16*)alloc(1024 * 256 * 2); bf16* whh2r = (bf16*)alloc(1024 * 256 * 2);
    bf16* emb2w = (bf16*)alloc(256 * 512 * 2);
    bf16* i2hb = (bf16*)alloc(128 * 256 * 2); bf16* h2hb16 = (bf16*)alloc(128 * 128 * 2);
    bf16* gwihb = (bf16*)alloc(384 * 256 * 2); bf16* gwhhb = (bf16*)alloc(384 * 128 * 2);
    bf16* genwb = (bf16*)alloc(1000 * 128 * 2);
    float* bc1f = (float*)alloc(1024 * 4); float* bc1r = (float*)alloc(1024 * 4);
    float* bc2f = (float*)alloc(1024 * 4); float* bc2r = (float*)alloc(1024 * 4);
    const int TB = T * 32;  // 4128
    bf16* x1 = (bf16*)alloc((size_t)TB * 512 * 2);
    float* xpf = (float*)alloc((size_t)TB * 1024 * 4);
    float* xpr = (float*)alloc((size_t)TB * 1024 * 4);
    bf16* hs_cat1 = (bf16*)alloc((size_t)TB * 512 * 2);
    float* emb1out = (float*)alloc((size_t)TB * 256 * 4);
    bf16* x2 = (bf16*)alloc((size_t)TB * 256 * 2);
    bf16* hs_cat2 = (bf16*)alloc((size_t)TB * 512 * 2);
    float* feats = (float*)alloc((size_t)TB * 256 * 4);
    bf16* featsb = (bf16*)alloc((size_t)TB * 256 * 2);
    float* fproj = (float*)alloc((size_t)TB * 128 * 4);
    float* hs_dec = (float*)alloc((size_t)STEPS * 32 * 128 * 4);
    int* cum = (int*)alloc(33 * 4);
    bf16* Agen = (bf16*)alloc((size_t)TOTAL * 128 * 2);

    auto cvt = [&](const float* s, bf16* d, int n) {
        k_f32_to_bf16<<<ceil_div(n, 256), 256, 0, stream>>>(s, d, n);
    };
    auto gemm = [&](const bf16* A, int lda, const bf16* B, int ldb, const float* bias,
                    float* C, int ldc, int M, int N, int K, int relu) {
        int mblocks = ceil_div(M / 16, 4);
        int ngroups = ceil_div(N, 64);
        k_gemm<<<mblocks * ngroups, 128, 0, stream>>>(A, lda, B, ldb, bias, C, ldc, M, N,
                                                      K, relu);
    };

    // ---- weight prep ----
    for (int i = 0; i < 7; ++i) {
        int n = cd[i][5] * Kpad[i];
        k_wprep<<<ceil_div(n, 256), 256, 0, stream>>>(CW[i], wk[i], cd[i][5], cd[i][2],
                                                      cd[i][6], cd[i][7], Kpad[i]);
    }
    k_ssprep<<<1, 64, 0, stream>>>(CB[0], nullptr, nullptr, nullptr, nullptr, csc[0], csh[0], 64, 0);
    k_ssprep<<<1, 128, 0, stream>>>(CB[1], nullptr, nullptr, nullptr, nullptr, csc[1], csh[1], 128, 0);
    k_ssprep<<<1, 256, 0, stream>>>(CB[2], BN2[0], BN2[1], BN2[2], BN2[3], csc[2], csh[2], 256, 1);
    k_ssprep<<<1, 256, 0, stream>>>(CB[3], nullptr, nullptr, nullptr, nullptr, csc[3], csh[3], 256, 0);
    k_ssprep<<<2, 256, 0, stream>>>(CB[4], BN4[0], BN4[1], BN4[2], BN4[3], csc[4], csh[4], 512, 1);
    k_ssprep<<<2, 256, 0, stream>>>(CB[5], nullptr, nullptr, nullptr, nullptr, csc[5], csh[5], 512, 0);
    k_ssprep<<<2, 256, 0, stream>>>(CB[6], BN6[0], BN6[1], BN6[2], BN6[3], csc[6], csh[6], 512, 1);
    cvt(L1F[0], wih1f, 1024 * 512); cvt(L1F[1], whh1f, 1024 * 256);
    cvt(L1R[0], wih1r, 1024 * 512); cvt(L1R[1], whh1r, 1024 * 256);
    cvt(EMB1W, emb1w, 256 * 512);
    cvt(L2F[0], wih2f, 1024 * 256); cvt(L2F[1], whh2f, 1024 * 256);
    cvt(L2R[0], wih2r, 1024 * 256); cvt(L2R[1], whh2r, 1024 * 256);
    cvt(EMB2W, emb2w, 256 * 512);
    cvt(I2HW, i2hb, 128 * 256); cvt(H2HW, h2hb16, 128 * 128);
    cvt(GWIH, gwihb, 384 * 256); cvt(GWHH, gwhhb, 384 * 128);
    cvt(GENW, genwb, 1000 * 128);
    k_vec_add<<<4, 256, 0, stream>>>(L1F[2], L1F[3], bc1f, 1024);
    k_vec_add<<<4, 256, 0, stream>>>(L1R[2], L1R[3], bc1r, 1024);
    k_vec_add<<<4, 256, 0, stream>>>(L2F[2], L2F[3], bc2f, 1024);
    k_vec_add<<<4, 256, 0, stream>>>(L2R[2], L2R[3], bc2r, 1024);
    cvt(IN_X, a0in, 32 * 32 * 512);  // NCHW C=1 == NHWC

    // ---- conv stack (NHWC bf16, ping-pong bufA/bufB) ----
    auto conv = [&](int i, const bf16* in, bf16* out) {
        int tiles = (32 * cd[i][3] * cd[i][4] / 16) * (cd[i][5] / 16);
        k_conv<<<ceil_div(tiles, 4), 128, 0, stream>>>(
            in, wk[i], csc[i], csh[i], out, cd[i][0], cd[i][1], cd[i][2], cd[i][3],
            cd[i][4], cd[i][5], cd[i][6], cd[i][7], cd[i][8], Kpad[i], Kreal[i]);
    };
    auto pool = [&](const bf16* in, bf16* out, int H, int W, int C, int OH, int OW,
                    int sh, int sw, int ph, int pw) {
        int n = 32 * OH * OW * C;
        k_pool<<<ceil_div(n, 256), 256, 0, stream>>>(in, out, H, W, C, OH, OW, sh, sw, ph, pw);
    };
    conv(0, a0in, bufA);
    pool(bufA, bufB, 32, 512, 64, 16, 256, 2, 2, 0, 0);
    conv(1, bufB, bufA);
    pool(bufA, bufB, 16, 256, 128, 8, 128, 2, 2, 0, 0);
    conv(2, bufB, bufA);
    conv(3, bufA, bufB);
    pool(bufB, bufA, 8, 128, 256, 4, 129, 2, 1, 0, 1);
    conv(4, bufA, bufB);
    conv(5, bufB, bufA);
    pool(bufA, bufB, 4, 129, 512, 2, 130, 2, 1, 0, 1);
    conv(6, bufB, bufA);  // [32,1,129,512] NHWC
    k_to_seq<<<ceil_div(TB * 512, 256), 256, 0, stream>>>(bufA, x1, T);

    // ---- BiLSTM layer 1 ----
    gemm(x1, 512, wih1f, 512, bc1f, xpf, 1024, TB, 1024, 512, 0);
    gemm(x1, 512, wih1r, 512, bc1r, xpr, 1024, TB, 1024, 512, 0);
    k_lstm<<<2, 1024, 0, stream>>>(xpf, xpr, whh1f, whh1r, hs_cat1, T);
    gemm(hs_cat1, 512, emb1w, 512, EMB1B, emb1out, 256, TB, 256, 512, 0);
    cvt(emb1out, x2, TB * 256);
    // ---- BiLSTM layer 2 ----
    gemm(x2, 256, wih2f, 256, bc2f, xpf, 1024, TB, 1024, 256, 0);
    gemm(x2, 256, wih2r, 256, bc2r, xpr, 1024, TB, 1024, 256, 0);
    k_lstm<<<2, 1024, 0, stream>>>(xpf, xpr, whh2f, whh2r, hs_cat2, T);
    gemm(hs_cat2, 512, emb2w, 512, EMB2B, feats, 256, TB, 256, 512, 0);
    cvt(feats, featsb, TB * 256);
    gemm(featsb, 256, i2hb, 256, nullptr, fproj, 128, TB, 128, 256, 0);

    // ---- attention-GRU decoder ----
    k_decoder<<<1, 1024, 0, stream>>>(fproj, feats, SCOREW, h2hb16, H2HB, gwihb, gwhhb,
                                      GBIH, GBHH, hs_dec, STEPS);

    // ---- ragged gather + generator ----
    k_cum<<<1, 32, 0, stream>>>(IN_LEN, cum);
    k_gather<<<ceil_div(TOTAL * 128, 256), 256, 0, stream>>>(hs_dec, cum, Agen, TOTAL);
    gemm(Agen, 128, genwb, 128, GENB, (float*)d_out, 1000, TOTAL, 1000, 128, 0);
}
